// PINN_59055800320683
// MI455X (gfx1250) — compile-verified, matrix operand order checked by reading, not run on hardware
//
#include <hip/hip_runtime.h>
#include <hip/hip_bf16.h>
#include <math.h>

#define B_      128
#define N_      128
#define D_      3
#define H_      128
#define DL_     5
#define P_      8128
#define PPAD_   8192
#define TILES_PER_BLOCK 8
#define DELTA2  0.0009f      // 0.03^2
#define GPARA_  0.25f
#define GAPARA_ 0.5f

typedef __attribute__((ext_vector_type(16))) _Float16 v16h;
typedef __attribute__((ext_vector_type(8)))  _Float16 v8h;
typedef __attribute__((ext_vector_type(8)))  float    v8f;

// ---- dynamic LDS partition (bytes) ------------------------------------
#define OFF_W1  0        // 128x128 f16  (layer-1 weights, n-major)  32768 B
#define OFF_H   32768    // 128x128 f16  (activations, row-major)    32768 B
#define OFF_W2  65536    //  16x128 f16  (layer-2 weights, padded)    4096 B
#define OFF_F   69632    // 128x4  f32   (per-row input features)     2048 B
#define OFF_RED 71680    //  16    f32   (block reduction slots)        64 B
#define OFF_CW  71744    // 648    f32   (W0 384 | b0 128 | b1 128 | b2 8)
#define SMEM_BYTES 74336

// ---- workspace partition (bytes) --------------------------------------
#define WS_ACCUM   0          // 2048 f32
#define WS_PI      8192       // 8192 i32
#define WS_PJ      40960      // 8192 i32
#define WS_PSIW1H  73728      // 16384 f16
#define WS_PSIW2H  106496     // 2048 f16 (padded 16x128)
#define WS_PHIW1H  110592     // 16384 f16
#define WS_PHIW2H  143360     // 2048 f16

__device__ __forceinline__ float gelu_exact(float v) {
    return 0.5f * v * (1.0f + erff(v * 0.70710678118654752440f));
}

// v_wmma_f32_16x16x32_f16 fragment loader. A and B use the symmetric layout:
// lane%16 -> M (A) / N (B); K split per ISA 7.12.2 16-bit A table:
//   elements 0..7  -> k = k0 + hi*8 + e
//   elements 8..15 -> k = k0 + 16 + hi*8 + (e-8)      (hi = lane/16)
// Row-major f16 LDS tile => two 16-byte ds_load_b128 per fragment per lane.
__device__ __forceinline__ v16h load_frag16(const _Float16* base, int row, int ld, int k0) {
    int hi = (threadIdx.x >> 4) & 1;
    const _Float16* p = base + row * ld + k0 + hi * 8;
    v8h lo = *(const v8h*)(p);
    v8h hh = *(const v8h*)(p + 16);
    v16h r;
#pragma unroll
    for (int e = 0; e < 8; ++e) { r[e] = lo[e]; r[e + 8] = hh[e]; }
    return r;
}

// -----------------------------------------------------------------------
// init: zero accumulators, build triu(k=1) pair tables, pre-convert both
// MLPs' layer-1/2 weights to f16 (W2 zero-padded to 16 output columns).
// -----------------------------------------------------------------------
__global__ __launch_bounds__(256) void pinn_init_kernel(
        const float* __restrict__ psiW1, const float* __restrict__ psiW2,
        const float* __restrict__ phiW1, const float* __restrict__ phiW2,
        float* __restrict__ accum, int* __restrict__ pi, int* __restrict__ pj,
        _Float16* __restrict__ psiW1h, _Float16* __restrict__ psiW2h,
        _Float16* __restrict__ phiW1h, _Float16* __restrict__ phiW2h) {
    int t = blockIdx.x * blockDim.x + threadIdx.x;
    if (t < B_ * 16) accum[t] = 0.0f;
    if (t < PPAD_) {
        if (t < P_) {
            int i = 0, rem = t;
            while (rem >= (N_ - 1 - i)) { rem -= (N_ - 1 - i); ++i; }
            pi[t] = i;
            pj[t] = i + 1 + rem;
        } else {
            pi[t] = 0;
            pj[t] = 1;
        }
    }
    if (t < H_ * H_) {
        psiW1h[t] = (_Float16)psiW1[t];
        phiW1h[t] = (_Float16)phiW1[t];
    }
    if (t < 16 * H_) {
        int n = t >> 7;
        psiW2h[t] = (n < DL_) ? (_Float16)psiW2[t] : (_Float16)0.0f;
        phiW2h[t] = (n < DL_) ? (_Float16)phiW2[t] : (_Float16)0.0f;
    }
}

// -----------------------------------------------------------------------
// stage pre-converted f16 weights + small constants into LDS
// -----------------------------------------------------------------------
__device__ __forceinline__ void stage_block(char* smem,
        const _Float16* __restrict__ W1h, const _Float16* __restrict__ W2h,
        const float* __restrict__ W0, const float* __restrict__ b0,
        const float* __restrict__ b1, const float* __restrict__ b2) {
    const int tid = threadIdx.x;
    uint4* dW1 = (uint4*)(smem + OFF_W1);
    const uint4* sW1 = (const uint4*)W1h;
    for (int idx = tid; idx < (H_ * H_) / 8; idx += 256) dW1[idx] = sW1[idx];
    uint4* dW2 = (uint4*)(smem + OFF_W2);
    const uint4* sW2 = (const uint4*)W2h;
    for (int idx = tid; idx < (16 * H_) / 8; idx += 256) dW2[idx] = sW2[idx];
    float* cw = (float*)(smem + OFF_CW);
    if (tid < 384) cw[tid] = W0[tid];
    if (tid < 128) { cw[384 + tid] = b0[tid]; cw[512 + tid] = b1[tid]; }
    if (tid < 8)   cw[640 + tid] = (tid < DL_) ? b2[tid] : 0.0f;
    float* sRed = (float*)(smem + OFF_RED);
    if (tid < 16) sRed[tid] = 0.0f;
}

// -----------------------------------------------------------------------
// shared MLP tail: layer1 (128x128, WMMA) + GELU + layer2 (128x16 padded,
// WMMA). Each wave owns rows [wave*16, wave*16+16) of sH. Result tile c2:
// lane n = tid&15, VGPR v -> m = wave*16 + (tid>>4&1)*8 + v.
// -----------------------------------------------------------------------
__device__ __forceinline__ v8f mlp_tail_wmma(_Float16* sH, const _Float16* sW1,
                                             const _Float16* sW2,
                                             const float* sB1) {
    const int tid = threadIdx.x;
    const int m0  = (tid >> 5) * 16;
    const int lr  = tid & 15;
    const int mh  = (tid >> 4) & 1;

    v16h a[4];
#pragma unroll
    for (int kt = 0; kt < 4; ++kt)
        a[kt] = load_frag16(sH, m0 + lr, H_, kt * 32);

    v8f acc[8];
#pragma unroll
    for (int nt = 0; nt < 8; ++nt) {
        v8f c = {};
#pragma unroll
        for (int kt = 0; kt < 4; ++kt) {
            v16h bf = load_frag16(sW1, nt * 16 + lr, H_, kt * 32);
            c = __builtin_amdgcn_wmma_f32_16x16x32_f16(false, a[kt], false, bf,
                                                       (short)0, c, false, false);
        }
        acc[nt] = c;
    }

    // bias + exact GELU, write f16 back to this wave's own rows
#pragma unroll
    for (int nt = 0; nt < 8; ++nt) {
        int n = nt * 16 + lr;
        float bias = sB1[n];
#pragma unroll
        for (int v = 0; v < 8; ++v) {
            int m = m0 + mh * 8 + v;
            sH[m * H_ + n] = (_Float16)gelu_exact(acc[nt][v] + bias);
        }
    }

    v8f c2 = {};
#pragma unroll
    for (int kt = 0; kt < 4; ++kt) {
        v16h a2 = load_frag16(sH, m0 + lr, H_, kt * 32);
        v16h b2 = load_frag16(sW2, lr, H_, kt * 32);
        c2 = __builtin_amdgcn_wmma_f32_16x16x32_f16(false, a2, false, b2,
                                                    (short)0, c2, false, false);
    }
    return c2;
}

// -----------------------------------------------------------------------
// psi branch: one block = 8 tiles x 128 pairs of one batch (weights staged
// once, reused across tiles). accum[b*16 + 0..4] = sum psi_out,
// [5] = sum r_soft, [6] = sum cos, [7] = sum gamma*r_soft (cusp)
// -----------------------------------------------------------------------
__global__ __launch_bounds__(256) void pinn_psi_kernel(
        const float* __restrict__ x, const int* __restrict__ spin,
        const float* __restrict__ W0, const float* __restrict__ b0,
        const _Float16* __restrict__ W1h, const float* __restrict__ b1,
        const _Float16* __restrict__ W2h, const float* __restrict__ b2,
        const int* __restrict__ pi, const int* __restrict__ pj,
        float* __restrict__ accum) {
    extern __shared__ __align__(16) char smem[];
    _Float16* sW1 = (_Float16*)(smem + OFF_W1);
    _Float16* sH  = (_Float16*)(smem + OFF_H);
    _Float16* sW2 = (_Float16*)(smem + OFF_W2);
    float*    sF  = (float*)(smem + OFF_F);
    float*    sRed = (float*)(smem + OFF_RED);
    float*    sW0 = (float*)(smem + OFF_CW);
    float*    sB0 = sW0 + 384;
    float*    sB1 = sW0 + 512;
    float*    sB2 = sW0 + 640;

    const int tid = threadIdx.x;
    const int b = blockIdx.x / TILES_PER_BLOCK;
    const int group = blockIdx.x % TILES_PER_BLOCK;

    stage_block(smem, W1h, W2h, W0, b0, b1, b2);

    for (int it = 0; it < TILES_PER_BLOCK; ++it) {
        const int tilebase = (group * TILES_PER_BLOCK + it) * 128;
        __syncthreads();   // prev tile's sH/sF consumers done; staging done (it==0)

        if (tid < 128) {
            int p = tilebase + tid;
            bool valid = (p < P_);
            int i = valid ? pi[p] : 0;
            int j = valid ? pj[p] : 1;
            const float* xi = x + (size_t)(b * N_ + i) * D_;
            const float* xj = x + (size_t)(b * N_ + j) * D_;
            float d0 = xi[0] - xj[0], d1 = xi[1] - xj[1], d2 = xi[2] - xj[2];
            float r2 = d0 * d0 + d1 * d1 + d2 * d2;
            float r  = sqrtf(r2 + DELTA2);
            float iv = 1.0f / (1.0f + r);
            float cr = cosf(r);
            sF[tid * 4 + 0] = r;
            sF[tid * 4 + 1] = iv;
            sF[tid * 4 + 2] = cr;
            if (valid) {
                float gamma = (spin[i] == spin[j]) ? GPARA_ : GAPARA_;
                atomicAdd(&sRed[5], r);
                atomicAdd(&sRed[6], cr);
                atomicAdd(&sRed[7], gamma * r);
            }
        }
        __syncthreads();

        // layer 0: 3 -> 128
        for (int idx = tid; idx < 128 * H_; idx += 256) {
            int row = idx >> 7, h = idx & 127;
            float v = sF[row * 4 + 0] * sW0[h * 3 + 0]
                    + sF[row * 4 + 1] * sW0[h * 3 + 1]
                    + sF[row * 4 + 2] * sW0[h * 3 + 2] + sB0[h];
            sH[row * H_ + h] = (_Float16)gelu_exact(v);
        }
        __syncthreads();

        v8f c2 = mlp_tail_wmma(sH, sW1, sW2, sB1);

        const int m0 = (tid >> 5) * 16;
        const int nl = tid & 15;
        const int mh = (tid >> 4) & 1;
        if (nl < DL_) {
            float bias2 = sB2[nl];
#pragma unroll
            for (int v = 0; v < 8; ++v) {
                int m = m0 + mh * 8 + v;
                if (tilebase + m < P_) atomicAdd(&sRed[nl], c2[v] + bias2);
            }
        }
    }
    __syncthreads();
    if (tid < 8) atomicAdd(&accum[b * 16 + tid], sRed[tid]);
}

// -----------------------------------------------------------------------
// phi branch: one block = one batch (N=128 particles = one tile)
// accum[b*16 + 8..12] = sum phi_out, [13] = sum x^2
// -----------------------------------------------------------------------
__global__ __launch_bounds__(256) void pinn_phi_kernel(
        const float* __restrict__ x,
        const float* __restrict__ W0, const float* __restrict__ b0,
        const _Float16* __restrict__ W1h, const float* __restrict__ b1,
        const _Float16* __restrict__ W2h, const float* __restrict__ b2,
        float* __restrict__ accum) {
    extern __shared__ __align__(16) char smem[];
    _Float16* sW1 = (_Float16*)(smem + OFF_W1);
    _Float16* sH  = (_Float16*)(smem + OFF_H);
    _Float16* sW2 = (_Float16*)(smem + OFF_W2);
    float*    sF  = (float*)(smem + OFF_F);
    float*    sRed = (float*)(smem + OFF_RED);
    float*    sW0 = (float*)(smem + OFF_CW);
    float*    sB0 = sW0 + 384;
    float*    sB1 = sW0 + 512;
    float*    sB2 = sW0 + 640;

    const int tid = threadIdx.x;
    const int b = blockIdx.x;

    stage_block(smem, W1h, W2h, W0, b0, b1, b2);
    __syncthreads();

    if (tid < N_) {
        const float* xr = x + (size_t)(b * N_ + tid) * D_;
        float f0 = xr[0], f1 = xr[1], f2 = xr[2];   // * sqrt(OMEGA) == 1
        sF[tid * 4 + 0] = f0;
        sF[tid * 4 + 1] = f1;
        sF[tid * 4 + 2] = f2;
        atomicAdd(&sRed[5], f0 * f0 + f1 * f1 + f2 * f2);
    }
    __syncthreads();

    for (int idx = tid; idx < 128 * H_; idx += 256) {
        int row = idx >> 7, h = idx & 127;
        float v = sF[row * 4 + 0] * sW0[h * 3 + 0]
                + sF[row * 4 + 1] * sW0[h * 3 + 1]
                + sF[row * 4 + 2] * sW0[h * 3 + 2] + sB0[h];
        sH[row * H_ + h] = (_Float16)gelu_exact(v);
    }
    __syncthreads();

    v8f c2 = mlp_tail_wmma(sH, sW1, sW2, sB1);

    {
        const int nl = tid & 15;
        if (nl < DL_) {
            float bias2 = sB2[nl];
#pragma unroll
            for (int v = 0; v < 8; ++v) atomicAdd(&sRed[nl], c2[v] + bias2);
        }
    }
    __syncthreads();
    if (tid < DL_) atomicAdd(&accum[b * 16 + 8 + tid], sRed[tid]);
    if (tid == 5)  atomicAdd(&accum[b * 16 + 13], sRed[5]);
}

// -----------------------------------------------------------------------
// finalize: means + rho head + cusp
// -----------------------------------------------------------------------
__global__ __launch_bounds__(128) void pinn_finalize_kernel(
        const float* __restrict__ accum, const float* __restrict__ rhoW,
        const float* __restrict__ rhob, float* __restrict__ out) {
    int b = threadIdx.x;
    if (b >= B_) return;
    const float* a = accum + b * 16;
    float rin[13];
#pragma unroll
    for (int n = 0; n < DL_; ++n) rin[n] = a[8 + n] * (1.0f / (float)N_);      // phi_mean
#pragma unroll
    for (int n = 0; n < DL_; ++n) rin[5 + n] = a[n] * (1.0f / (float)P_);      // psi_mean
    rin[10] = a[13] * (1.0f / (float)(N_ * D_));                               // r2_mean
    rin[11] = a[5] * (1.0f / (float)P_);                                       // mean_r
    rin[12] = a[6] * (1.0f / (float)P_) * 0.5f;                                // cos1
    float o = rhob[0] + a[7];                                                  // bias + cusp
#pragma unroll
    for (int k = 0; k < 13; ++k) o += rhoW[k] * rin[k];
    out[b] = o;
}

// -----------------------------------------------------------------------
extern "C" void kernel_launch(void* const* d_in, const int* in_sizes, int n_in,
                              void* d_out, int out_size, void* d_ws, size_t ws_size,
                              hipStream_t stream) {
    const float* x     = (const float*)d_in[0];
    const int*   spin  = (const int*)d_in[1];
    const float* phiW0 = (const float*)d_in[2];
    const float* phib0 = (const float*)d_in[3];
    const float* phiW1 = (const float*)d_in[4];
    const float* phib1 = (const float*)d_in[5];
    const float* phiW2 = (const float*)d_in[6];
    const float* phib2 = (const float*)d_in[7];
    const float* psiW0 = (const float*)d_in[8];
    const float* psib0 = (const float*)d_in[9];
    const float* psiW1 = (const float*)d_in[10];
    const float* psib1 = (const float*)d_in[11];
    const float* psiW2 = (const float*)d_in[12];
    const float* psib2 = (const float*)d_in[13];
    const float* rhoW  = (const float*)d_in[14];
    const float* rhob  = (const float*)d_in[15];

    char* ws = (char*)d_ws;
    float*    accum  = (float*)(ws + WS_ACCUM);
    int*      pi     = (int*)(ws + WS_PI);
    int*      pj     = (int*)(ws + WS_PJ);
    _Float16* psiW1h = (_Float16*)(ws + WS_PSIW1H);
    _Float16* psiW2h = (_Float16*)(ws + WS_PSIW2H);
    _Float16* phiW1h = (_Float16*)(ws + WS_PHIW1H);
    _Float16* phiW2h = (_Float16*)(ws + WS_PHIW2H);

    pinn_init_kernel<<<(H_ * H_) / 256, 256, 0, stream>>>(
        psiW1, psiW2, phiW1, phiW2, accum, pi, pj,
        psiW1h, psiW2h, phiW1h, phiW2h);

    pinn_psi_kernel<<<B_ * TILES_PER_BLOCK, 256, SMEM_BYTES, stream>>>(
        x, spin, psiW0, psib0, psiW1h, psib1, psiW2h, psib2, pi, pj, accum);

    pinn_phi_kernel<<<B_, 256, SMEM_BYTES, stream>>>(
        x, phiW0, phib0, phiW1h, phib1, phiW2h, phib2, accum);

    pinn_finalize_kernel<<<1, 128, 0, stream>>>(accum, rhoW, rhob, (float*)d_out);
}
